// CDNSR_77025943486941
// MI455X (gfx1250) — matrix operand order repaired and checked by reading, not compiled, not run on hardware
//
#include <hip/hip_runtime.h>
#include <hip/hip_bf16.h>
#include <math.h>

typedef __attribute__((ext_vector_type(16))) _Float16 v16h;
typedef __attribute__((ext_vector_type(8)))  float    v8f;

// ---------------------------------------------------------------------------
// Generic implicit-GEMM conv on WMMA 16x16x32 (f16 in, f32 accum), templated
// on the kernel window so B-side K decomposition uses constant div/mod.
// A operand is contiguous in K (w[oc*Ktot + k]). Each wave computes a
// 16(M) x 32(N) tile -> 2 v_wmma per 32-deep K chunk with A reuse.
// ---------------------------------------------------------------------------
template <int TKH, int TKW>
__global__ __launch_bounds__(128)
void conv_k_wmma_kernel(const float* __restrict__ in, const float* __restrict__ w,
                        const float* __restrict__ bias, float* __restrict__ out,
                        const float* __restrict__ res,
                        int N, int CinG, int H, int W,
                        int CoutG, int Hout, int Wout,
                        int stride, int pad, int tilesM,
                        int CinTot, int in_off, int CoutTot, int out_off,
                        int CresTot, int res_off, int act, int shuf)
{
    constexpr int KK = TKH * TKW;
    const int g      = blockIdx.x / tilesM;
    const int tileM  = blockIdx.x % tilesM;
    const int wave   = threadIdx.x >> 5;
    const int lane   = threadIdx.x & 31;
    const int NPix   = N * Hout * Wout;
    const int tileN0 = (blockIdx.y * 4 + wave) * 2;
    if (tileN0 * 16 >= NPix) return;

    const int Ktot = CinG * KK;
    const int HW   = H * W;
    const int HWo  = Hout * Wout;

    const int mA    = lane & 15;
    const int koffA = (lane >> 4) * 8;
    const int mAc   = (mA < CoutG) ? mA : (CoutG - 1);
    const float* arow = w + (size_t)(g * CoutG + tileM * 16 + mAc) * Ktot;
    const bool mAv  = (tileM * 16 + mA) < CoutG;

    const int nB    = lane & 15;
    const int koffB = (lane >> 4) * 16;
    int bB[2], oyB[2], oxB[2];
    bool ntv[2];
#pragma unroll
    for (int t = 0; t < 2; ++t) {
        int pix = (tileN0 + t) * 16 + nB;
        ntv[t] = pix < NPix;
        if (pix >= NPix) pix = NPix - 1;
        bB[t]  = pix / HWo;
        int r  = pix - bB[t] * HWo;
        oyB[t] = r / Wout;
        oxB[t] = r - oyB[t] * Wout;
    }
    const float* inb[2];
#pragma unroll
    for (int t = 0; t < 2; ++t)
        inb[t] = in + ((size_t)bB[t] * CinTot + in_off + g * CinG) * HW;

    v8f acc0 = {}, acc1 = {};
    for (int k0 = 0; k0 < Ktot; k0 += 32) {
        v16h va, vb0, vb1;
#pragma unroll
        for (int v = 0; v < 8; ++v) {
            const int kbA = (v < 4) ? (koffA + 2 * v) : (16 + koffA + 2 * (v - 4));
#pragma unroll
            for (int e = 0; e < 2; ++e) {
                const int k = k0 + kbA + e;
                va[2 * v + e] = (_Float16)((mAv && k < Ktot) ? arow[k] : 0.0f);
            }
#pragma unroll
            for (int e = 0; e < 2; ++e) {
                const int k  = k0 + kbA + e;
                const int ic = k / KK;
                const int rr = k - ic * KK;
                const int ky = rr / TKW;
                const int kx = rr - ky * TKW;
                float x0 = 0.0f, x1 = 0.0f;
                if (k < Ktot) {
#pragma unroll
                    for (int t = 0; t < 2; ++t) {
                        const int iy = oyB[t] * stride - pad + ky;
                        const int ix = oxB[t] * stride - pad + kx;
                        float xv = 0.0f;
                        if (ntv[t] && iy >= 0 && iy < H && ix >= 0 && ix < W)
                            xv = inb[t][(size_t)ic * HW + (size_t)iy * W + ix];
                        if (t == 0) x0 = xv; else x1 = xv;
                    }
                }
                vb0[2 * v + e] = (_Float16)x0;
                vb1[2 * v + e] = (_Float16)x1;
            }
        }
        acc0 = __builtin_amdgcn_wmma_f32_16x16x32_f16(false, va, false, vb0, (short)0, acc0, false, false);
        acc1 = __builtin_amdgcn_wmma_f32_16x16x32_f16(false, va, false, vb1, (short)0, acc1, false, false);
    }

    const int mhi = (lane >> 4) * 8;
#pragma unroll
    for (int t = 0; t < 2; ++t) {
        if (!ntv[t]) continue;
        const v8f a = (t == 0) ? acc0 : acc1;
#pragma unroll
        for (int r = 0; r < 8; ++r) {
            const int ocg = tileM * 16 + r + mhi;
            if (ocg >= CoutG) continue;
            float v = a[r] + bias[g * CoutG + ocg];
            if (res != nullptr)
                v += res[((size_t)bB[t] * CresTot + res_off + g * CoutG + ocg) * HWo
                         + (size_t)oyB[t] * Wout + oxB[t]];
            if (act == 1)      v = fmaxf(v, 0.0f);
            else if (act == 2) v = (v >= 0.0f) ? v : 0.1f * v;

            if (!shuf) {
                out[((size_t)bB[t] * CoutTot + out_off + g * CoutG + ocg) * HWo
                    + (size_t)oyB[t] * Wout + oxB[t]] = v;
            } else {
                const int occ = g * CoutG + ocg;
                const int c4  = occ >> 2;
                const int rb  = occ & 3;
                out[((size_t)bB[t] * CoutTot + c4) * (size_t)(4 * HWo)
                    + (size_t)(2 * oyB[t] + (rb >> 1)) * (2 * Wout)
                    + (2 * oxB[t] + (rb & 1))] = v;
            }
        }
    }
}

// ---------------------------------------------------------------------------
// Specialized 1x1 conv, templated on HW (channel stride) so all B gathers use
// the load instruction's immediate offset field (i*THW*4 fits 24-bit signed):
// no per-element 64-bit address adds, no xcnt churn. 32(M) x 32(N) register
// tile per wave, 4 v_wmma per 32-deep K chunk, vectorized A, prefetch.
// ---------------------------------------------------------------------------
template <int THW>
__global__ __launch_bounds__(128)
void conv1x1_wmma_kernel(const float* __restrict__ in, const float* __restrict__ w,
                         const float* __restrict__ bias, float* __restrict__ out,
                         const float* __restrict__ res,
                         int N, int Cin, int Cout,
                         int CinTot, int in_off, int CoutTot, int out_off,
                         int CresTot, int res_off, int act)
{
    const int tileM2 = blockIdx.x;
    const int wave   = threadIdx.x >> 5;
    const int lane   = threadIdx.x & 31;
    const int NPix   = N * THW;
    const int tileN0 = (blockIdx.y * 4 + wave) * 2;
    if (tileN0 * 16 >= NPix) return;

    const int mA    = lane & 15;
    const int koffA = (lane >> 4) * 8;
    const int nB    = lane & 15;
    const int koffB = (lane >> 4) * 8;

    const float* bbase[2];
    int bIdx[2], hwIdx[2];
    bool ntv[2];
#pragma unroll
    for (int t = 0; t < 2; ++t) {
        int pix = (tileN0 + t) * 16 + nB;
        ntv[t] = pix < NPix;
        if (pix >= NPix) pix = NPix - 1;
        const int b  = pix / THW;
        const int hw = pix - b * THW;
        bIdx[t] = b; hwIdx[t] = hw;
        bbase[t] = in + ((size_t)b * CinTot + in_off + koffB) * THW + hw;
    }

    const int oc0  = tileM2 * 32 + mA;
    const int oc1  = oc0 + 16;
    const int oc0c = (oc0 < Cout) ? oc0 : (Cout - 1);
    const int oc1c = (oc1 < Cout) ? oc1 : (Cout - 1);
    const float* arow0 = w + (size_t)oc0c * Cin + koffA;
    const float* arow1 = w + (size_t)oc1c * Cin + koffA;

    v8f acc00 = {}, acc01 = {}, acc10 = {}, acc11 = {};
    const int Kmain = Cin & ~31;

    for (int k0 = 0; k0 < Kmain; k0 += 32) {
        if (k0 + 32 < Cin) {
            __builtin_prefetch(arow0 + k0 + 32, 0, 1);
            __builtin_prefetch(bbase[0] + (size_t)(k0 + 32) * THW, 0, 1);
        }
        v16h a0, a1, b0, b1;
        {
            float t0[16], t1[16];
            *(float4*)(t0 + 0)  = *(const float4*)(arow0 + k0 + 0);
            *(float4*)(t0 + 4)  = *(const float4*)(arow0 + k0 + 4);
            *(float4*)(t0 + 8)  = *(const float4*)(arow0 + k0 + 16);
            *(float4*)(t0 + 12) = *(const float4*)(arow0 + k0 + 20);
            *(float4*)(t1 + 0)  = *(const float4*)(arow1 + k0 + 0);
            *(float4*)(t1 + 4)  = *(const float4*)(arow1 + k0 + 4);
            *(float4*)(t1 + 8)  = *(const float4*)(arow1 + k0 + 16);
            *(float4*)(t1 + 12) = *(const float4*)(arow1 + k0 + 20);
#pragma unroll
            for (int i = 0; i < 16; ++i) { a0[i] = (_Float16)t0[i]; a1[i] = (_Float16)t1[i]; }
        }
        {
            const float* p0 = bbase[0] + (size_t)k0 * THW;
            const float* p1 = bbase[1] + (size_t)k0 * THW;
#pragma unroll
            for (int i = 0; i < 16; ++i) {
                b0[i] = (_Float16)p0[(i < 8 ? i : i + 8) * THW];   // constant offsets -> load imm field
                b1[i] = (_Float16)p1[(i < 8 ? i : i + 8) * THW];
            }
        }
        acc00 = __builtin_amdgcn_wmma_f32_16x16x32_f16(false, a0, false, b0, (short)0, acc00, false, false);
        acc10 = __builtin_amdgcn_wmma_f32_16x16x32_f16(false, a1, false, b0, (short)0, acc10, false, false);
        acc01 = __builtin_amdgcn_wmma_f32_16x16x32_f16(false, a0, false, b1, (short)0, acc01, false, false);
        acc11 = __builtin_amdgcn_wmma_f32_16x16x32_f16(false, a1, false, b1, (short)0, acc11, false, false);
    }

    if (Kmain < Cin) {
        v16h a0, a1, b0, b1;
#pragma unroll
        for (int i = 0; i < 16; ++i) {
            const int kk = (i < 8) ? (Kmain + i) : (Kmain + 16 + (i - 8));
            const bool kv = kk < Cin;
            a0[i] = (_Float16)(kv ? arow0[kk] : 0.0f);
            a1[i] = (_Float16)(kv ? arow1[kk] : 0.0f);
            const int kb = kk;
            const bool kvb = (kb + koffB) < Cin;
            b0[i] = (_Float16)(kvb ? bbase[0][(size_t)kb * THW] : 0.0f);
            b1[i] = (_Float16)(kvb ? bbase[1][(size_t)kb * THW] : 0.0f);
        }
        acc00 = __builtin_amdgcn_wmma_f32_16x16x32_f16(false, a0, false, b0, (short)0, acc00, false, false);
        acc10 = __builtin_amdgcn_wmma_f32_16x16x32_f16(false, a1, false, b0, (short)0, acc10, false, false);
        acc01 = __builtin_amdgcn_wmma_f32_16x16x32_f16(false, a0, false, b1, (short)0, acc01, false, false);
        acc11 = __builtin_amdgcn_wmma_f32_16x16x32_f16(false, a1, false, b1, (short)0, acc11, false, false);
    }

    const int mhi = (lane >> 4) * 8;
#pragma unroll
    for (int mt = 0; mt < 2; ++mt) {
#pragma unroll
        for (int nt = 0; nt < 2; ++nt) {
            if (!ntv[nt]) continue;
            const v8f a = (mt == 0) ? (nt == 0 ? acc00 : acc01)
                                    : (nt == 0 ? acc10 : acc11);
#pragma unroll
            for (int r = 0; r < 8; ++r) {
                const int ocg = tileM2 * 32 + mt * 16 + r + mhi;
                if (ocg >= Cout) continue;
                float v = a[r] + bias[ocg];
                if (res != nullptr)
                    v += res[((size_t)bIdx[nt] * CresTot + res_off + ocg) * THW + hwIdx[nt]];
                if (act == 1)      v = fmaxf(v, 0.0f);
                else if (act == 2) v = (v >= 0.0f) ? v : 0.1f * v;
                out[((size_t)bIdx[nt] * CoutTot + out_off + ocg) * THW + hwIdx[nt]] = v;
            }
        }
    }
}

__global__ void copy_ch_kernel(const float* __restrict__ src, float* __restrict__ dst,
                               int total, int C, int HW,
                               int CsT, int soff, int CdT, int doff)
{
    int i = blockIdx.x * blockDim.x + threadIdx.x;
    if (i >= total) return;
    int hw = i % HW;
    int t  = i / HW;
    int c  = t % C;
    int n  = t / C;
    dst[((size_t)n * CdT + doff + c) * HW + hw] =
        src[((size_t)n * CsT + soff + c) * HW + hw];
}

__device__ __forceinline__ float urand_hash(unsigned idx)
{
    unsigned x = idx * 747796405u + 2891336453u;
    x = ((x >> ((x >> 28) + 4u)) ^ x) * 277803737u;
    x = (x >> 22) ^ x;
    float u = ((float)x + 0.5f) * (1.0f / 4294967296.0f);
    return fminf(fmaxf(u, 1e-6f), 1.0f - 1e-6f);
}

__global__ void cls_head_kernel(const float* __restrict__ h,
                                const float* __restrict__ fcw,
                                const float* __restrict__ fcb,
                                float* __restrict__ p_ws, float* __restrict__ p_out)
{
    const int b = blockIdx.x;
    const int t = threadIdx.x;
    __shared__ float mean[32];
    float s = 0.0f;
    const float* base = h + ((size_t)b * 32 + t) * 576;
    for (int i = 0; i < 576; ++i) s += base[i];
    mean[t] = s * (1.0f / 576.0f);
    __syncthreads();
    if (t == 0) {
        float l[3];
        for (int j = 0; j < 3; ++j) {
            float v = fcb[j];
            for (int c = 0; c < 32; ++c) v += mean[c] * fcw[j * 32 + c];
            float u = urand_hash((unsigned)(b * 3 + j) * 2654435761u + 7u);
            l[j] = v + (-logf(-logf(u)));
        }
        float mx = fmaxf(l[0], fmaxf(l[1], l[2]));
        float e0 = expf(l[0] - mx), e1 = expf(l[1] - mx), e2 = expf(l[2] - mx);
        float inv = 1.0f / (e0 + e1 + e2);
        p_ws[b * 3 + 0] = e0 * inv; p_ws[b * 3 + 1] = e1 * inv; p_ws[b * 3 + 2] = e2 * inv;
        p_out[b * 3 + 0] = e0 * inv; p_out[b * 3 + 1] = e1 * inv; p_out[b * 3 + 2] = e2 * inv;
    }
}

__global__ void wsum_kernel(float* __restrict__ dout, const float* __restrict__ p,
                            long long S, int perB)
{
    long long i = (long long)blockIdx.x * blockDim.x + threadIdx.x;
    if (i >= S) return;
    int b = (int)(i / perB);
    dout[i] = dout[S + i]     * p[b * 3 + 0]
            + dout[2 * S + i] * p[b * 3 + 1]
            + dout[3 * S + i] * p[b * 3 + 2];
}

// ---------------------------------------------------------------------------
extern "C" void kernel_launch(void* const* d_in, const int* in_sizes, int n_in,
                              void* d_out, int out_size, void* d_ws, size_t ws_size,
                              hipStream_t stream)
{
    (void)in_sizes; (void)n_in; (void)out_size; (void)ws_size;
    const int Nb = 8;
    const float* x = (const float*)d_in[0];
    auto P = [&](int i) { return (const float*)d_in[i]; };

    float* Wks = (float*)d_ws;
    size_t off = 0;
    auto alloc = [&](size_t n) { float* q = Wks + off; off += n; return q; };
    const size_t px96 = (size_t)Nb * 96 * 96;
    float* A   = alloc(256 * px96);
    float* Bc  = alloc(256 * px96);
    float* Q0  = alloc(64 * px96);
    float* Q1  = alloc(64 * px96);
    float* Q2  = alloc(64 * px96);
    float* T1  = alloc(64 * px96);
    float* T2  = alloc(64 * px96);
    float* S1  = alloc(64 * (size_t)Nb * 192 * 192);
    float* S2  = alloc(64 * (size_t)Nb * 384 * 384);
    float* CL1 = alloc(128 * (size_t)Nb * 24 * 24);
    float* CL2 = alloc(128 * (size_t)Nb * 24 * 24);
    float* CL3 = alloc(32 * (size_t)Nb * 24 * 24);
    float* Pb  = alloc(32);

    auto conv = [&](const float* inp, int CinTot, int in_off,
                    const float* w, const float* bias,
                    float* outp, int CoutTot, int out_off,
                    const float* res, int CresTot, int res_off,
                    int CinG, int H, int Wd, int CoutG,
                    int KH, int stride, int pad, int groups, int act, int shuf) {
        const int Hout = (H + 2 * pad - KH) / stride + 1;
        const int Wout = (Wd + 2 * pad - KH) / stride + 1;
        const long long NPix = (long long)Nb * Hout * Wout;
        const int HW = H * Wd;
        if (KH == 1 && groups == 1 && stride == 1 && pad == 0 && !shuf
            && (HW == 9216 || HW == 576)) {
            const int tilesM2 = (CoutG + 31) / 32;
            const int tilesN  = (int)((NPix + 15) / 16);
            dim3 grid(tilesM2, (tilesN + 7) / 8);
            if (HW == 9216)
                conv1x1_wmma_kernel<9216><<<grid, 128, 0, stream>>>(
                    inp, w, bias, outp, res, Nb, CinG, CoutG,
                    CinTot, in_off, CoutTot, out_off, CresTot, res_off, act);
            else
                conv1x1_wmma_kernel<576><<<grid, 128, 0, stream>>>(
                    inp, w, bias, outp, res, Nb, CinG, CoutG,
                    CinTot, in_off, CoutTot, out_off, CresTot, res_off, act);
        } else {
            const int tilesM = (CoutG + 15) / 16;
            const int tilesN = (int)((NPix + 15) / 16);
            dim3 grid(tilesM * groups, (tilesN + 7) / 8);
            if (KH == 3)
                conv_k_wmma_kernel<3, 3><<<grid, 128, 0, stream>>>(
                    inp, w, bias, outp, res,
                    Nb, CinG, H, Wd, CoutG, Hout, Wout,
                    stride, pad, tilesM,
                    CinTot, in_off, CoutTot, out_off, CresTot, res_off, act, shuf);
            else if (KH == 4)
                conv_k_wmma_kernel<4, 4><<<grid, 128, 0, stream>>>(
                    inp, w, bias, outp, res,
                    Nb, CinG, H, Wd, CoutG, Hout, Wout,
                    stride, pad, tilesM,
                    CinTot, in_off, CoutTot, out_off, CresTot, res_off, act, shuf);
            else
                conv_k_wmma_kernel<1, 1><<<grid, 128, 0, stream>>>(
                    inp, w, bias, outp, res,
                    Nb, CinG, H, Wd, CoutG, Hout, Wout,
                    stride, pad, tilesM,
                    CinTot, in_off, CoutTot, out_off, CresTot, res_off, act, shuf);
        }
    };

    auto copy_ch = [&](const float* src, int CsT, int soff,
                       float* dst, int CdT, int doff, int C) {
        const int HW = 96 * 96;
        const long long tot = (long long)Nb * C * HW;
        copy_ch_kernel<<<(int)((tot + 255) / 256), 256, 0, stream>>>(
            src, dst, (int)tot, C, HW, CsT, soff, CdT, doff);
    };

    auto run_net = [&](int base, int nf, float* outdst) {
        const int nfg = nf / 4;
        conv(x, 3, 0, P(base + 0), P(base + 1), A, 4 * nf, 0,
             nullptr, 0, 0, 3, 96, 96, nf, 3, 1, 1, 1, 0, 0);        // entry

        const float* ocur = A; int ocurT = 4 * nf; int ocurOff = 0;
        for (int blk = 0; blk < 3; ++blk) {
            const int bb = base + 14 + 12 * blk;
            const float *ew1 = P(bb + 0), *eb1 = P(bb + 1);
            const float *ew2 = P(bb + 2), *eb2 = P(bb + 3);
            const float *ew3 = P(bb + 4), *eb3 = P(bb + 5);

            copy_ch(ocur, ocurT, ocurOff, Bc, 4 * nf, 0, nf);

            const float* icur = ocur; int icurT = ocurT; int icurOff = ocurOff;
            float* innerO[2] = { Q1, Q2 };
            for (int j = 0; j < 3; ++j) {
                conv(icur, icurT, icurOff, ew1, eb1, T1, nf, 0,
                     nullptr, 0, 0, nfg, 96, 96, nfg, 3, 1, 1, 4, 1, 0);
                conv(T1, nf, 0, ew2, eb2, T2, nf, 0,
                     nullptr, 0, 0, nfg, 96, 96, nfg, 3, 1, 1, 4, 1, 0);
                conv(T2, nf, 0, ew3, eb3, Bc, 4 * nf, (j + 1) * nf,
                     icur, icurT, icurOff, nf, 96, 96, nf, 1, 1, 0, 1, 1, 0);
                const float* cw = P(bb + 6 + 2 * j);
                const float* cb = P(bb + 7 + 2 * j);
                float* dst; int dstT; int dstOff;
                if (j < 2) { dst = innerO[j]; dstT = nf; dstOff = 0; }
                else       { dst = A; dstT = 4 * nf; dstOff = (blk + 1) * nf; }
                conv(Bc, 4 * nf, 0, cw, cb, dst, dstT, dstOff,
                     nullptr, 0, 0, (j + 2) * nf, 96, 96, nf, 1, 1, 0, 1, 1, 0);
                icur = dst; icurT = dstT; icurOff = dstOff;
            }
            conv(A, 4 * nf, 0, P(base + 8 + 2 * blk), P(base + 9 + 2 * blk),
                 Q0, nf, 0, nullptr, 0, 0, (blk + 2) * nf, 96, 96, nf, 1, 1, 0, 1, 1, 0);
            ocur = Q0; ocurT = nf; ocurOff = 0;
        }
        conv(Q0, nf, 0, P(base + 4), P(base + 5), S1, nf, 0,
             nullptr, 0, 0, nfg, 96, 96, nf, 3, 1, 1, 4, 1, 1);      // up1+shuffle
        conv(S1, nf, 0, P(base + 6), P(base + 7), S2, nf, 0,
             nullptr, 0, 0, nfg, 192, 192, nf, 3, 1, 1, 4, 1, 1);    // up2+shuffle
        conv(S2, nf, 0, P(base + 2), P(base + 3), outdst, 3, 0,
             nullptr, 0, 0, nf, 384, 384, 3, 3, 1, 1, 1, 0, 0);      // exit
    };

    // classifier
    conv(x, 3, 0, P(1), P(2), CL1, 128, 0, nullptr, 0, 0,
         3, 96, 96, 128, 4, 4, 0, 1, 2, 0);
    conv(CL1, 128, 0, P(3), P(4), CL2, 128, 0, nullptr, 0, 0,
         128, 24, 24, 128, 1, 1, 0, 1, 2, 0);
    conv(CL2, 128, 0, P(5), P(6), CL1, 128, 0, nullptr, 0, 0,
         128, 24, 24, 128, 1, 1, 0, 1, 2, 0);
    conv(CL1, 128, 0, P(7), P(8), CL2, 128, 0, nullptr, 0, 0,
         128, 24, 24, 128, 1, 1, 0, 1, 2, 0);
    conv(CL2, 128, 0, P(9), P(10), CL3, 32, 0, nullptr, 0, 0,
         128, 24, 24, 32, 1, 1, 0, 1, 0, 0);

    float* fout = (float*)d_out;
    const long long S = (long long)Nb * 3 * 384 * 384;
    cls_head_kernel<<<Nb, 32, 0, stream>>>(CL3, P(11), P(12), Pb, fout + 4 * S);

    run_net(13,  36, fout + S);
    run_net(63,  52, fout + 2 * S);
    run_net(113, 64, fout + 3 * S);

    wsum_kernel<<<(int)((S + 255) / 256), 256, 0, stream>>>(fout, Pb, S, 3 * 384 * 384);
}
